// tcn_transformer_73787538145507
// MI455X (gfx1250) — compile-verified, matrix operand order checked by reading, not run on hardware
//
#include <hip/hip_runtime.h>
#include <hip/hip_bf16.h>
#include <math.h>

typedef __attribute__((ext_vector_type(16))) _Float16     v16h;
typedef __attribute__((ext_vector_type(8)))  float        v8f;
typedef __attribute__((ext_vector_type(4)))  unsigned int v4u;
typedef __attribute__((ext_vector_type(4)))  int          v4i;
typedef __attribute__((address_space(1)))    v4i          gv4i;   // global int4
typedef __attribute__((address_space(3)))    v4i          lv4i;   // LDS int4

#define NB   256
#define CI   128
#define TT   64
#define VV   25
#define VG   5      // v-values per block
#define HH   4
#define DD   32
#define OUTC 128

// ---- workspace layout (bytes) ----
#define WS_WQKV   0          // 384*128 f16 = 98304
#define WS_WPROJ  98304      // 128*128 f16 = 32768  (contiguous with WQKV)
#define WS_BQKV   131072     // 384 f32
#define WS_BPROJ  132608     // 128 f32
#define WS_RSCALE 133120     // 128 f32

// ---- LDS layout (bytes) ----
#define SM_XS     0          // [64][128] f16   = 16384
#define SM_WQKV   16384      // [384][128] f16  = 98304
#define SM_WPROJ  114688     // [128][128] f16  = 32768  (contiguous with WQKV)
#define SM_Q      147456     // [4][64][32] f16 = 16384
#define SM_K      163840     // [4][64][32] f16 = 16384
#define SM_VT     180224     // [4][32][64] f16 = 16384
#define SM_S      196608     // [4][64][64] f32 = 65536 (attn f16 packed in-place)
#define SM_AO     262144     // [64][128] f16   = 16384
#define SM_TOTAL  278528

#define USE_ASYNC_LDS (__has_builtin(__builtin_amdgcn_global_load_async_to_lds_b128) && \
                       __has_builtin(__builtin_amdgcn_s_wait_asynccnt))

// 16x32 f16 A-fragment (or B^T row-fragment) from LDS, row-major.
// p = &row[kbase + lhi*8]; halves 0..7 and 16..23 -> two aligned b128 loads.
__device__ inline v16h frag_ld(const _Float16* p) {
    union { v16h h; v4u u[2]; } r;
    const v4u* q = (const v4u*)p;       // 16B aligned by construction
    r.u[0] = q[0];                      // dwords 0..3  (halves 0..7)
    r.u[1] = q[2];                      // dwords 8..11 (halves 16..23)
    return r.h;
}

// Fold input-BN into qk/v weights+bias, output-BN into proj weights/bias and
// residual scale:  w' = W*inv_c ; b' = b + W@(beta - mean*inv_c)
// out = (proj@a)*inv_o + [inv_o*(pb-mean_o)+beta_o] + inv_o * x_in
__global__ __launch_bounds__(256) void fold_kernel(
    const float* __restrict__ bn_gamma, const float* __restrict__ bn_beta,
    const float* __restrict__ bn_mean,  const float* __restrict__ bn_var,
    const float* __restrict__ qk_w,     const float* __restrict__ qk_b,
    const float* __restrict__ v_w,      const float* __restrict__ v_b,
    const float* __restrict__ proj_w,   const float* __restrict__ proj_b,
    const float* __restrict__ bno_gamma,const float* __restrict__ bno_beta,
    const float* __restrict__ bno_mean, const float* __restrict__ bno_var,
    _Float16* __restrict__ wqkv, _Float16* __restrict__ wproj,
    float* __restrict__ bqkv, float* __restrict__ bproj, float* __restrict__ rscale)
{
    int r = blockIdx.x * 256 + threadIdx.x;
    if (r < 384) {
        const float* wsrc = (r < 256) ? (qk_w + (size_t)r * CI) : (v_w + (size_t)(r - 256) * CI);
        float acc = (r < 256) ? qk_b[r] : v_b[r - 256];
        for (int c = 0; c < CI; c++) {
            float inv = bn_gamma[c] * rsqrtf(bn_var[c] + 1e-5f);
            float w = wsrc[c];
            wqkv[r * CI + c] = (_Float16)(w * inv);
            acc += w * (bn_beta[c] - bn_mean[c] * inv);
        }
        bqkv[r] = acc;
    } else if (r < 512) {
        int o = r - 384;
        float inv = bno_gamma[o] * rsqrtf(bno_var[o] + 1e-5f);
        for (int c = 0; c < OUTC; c++)
            wproj[o * OUTC + c] = (_Float16)(proj_w[o * OUTC + c] * inv);
        bproj[o]  = inv * (proj_b[o] - bno_mean[o]) + bno_beta[o];
        rscale[o] = inv;
    }
}

__global__ __launch_bounds__(256) void attn_fused(
    const float* __restrict__ x, const float* __restrict__ logit_scale,
    const float* __restrict__ rel,
    const _Float16* __restrict__ gwqkv,      // wqkv||wproj contiguous, 131072 B
    const float* __restrict__ bqkv, const float* __restrict__ bproj,
    const float* __restrict__ rscale, float* __restrict__ out)
{
    extern __shared__ char smem[];
    _Float16* xs    = (_Float16*)(smem + SM_XS);
    _Float16* wqkv  = (_Float16*)(smem + SM_WQKV);
    _Float16* wproj = (_Float16*)(smem + SM_WPROJ);
    _Float16* qbuf  = (_Float16*)(smem + SM_Q);
    _Float16* kbuf  = (_Float16*)(smem + SM_K);
    _Float16* vbufT = (_Float16*)(smem + SM_VT);
    float*    sbuf  = (float*)   (smem + SM_S);
    _Float16* ao    = (_Float16*)(smem + SM_AO);
    _Float16* smh   = (_Float16*)smem;

    const int tid  = threadIdx.x;
    const int wave = tid >> 5;
    const int lane = tid & 31;
    const int lm   = lane & 15;
    const int lhi  = lane >> 4;
    const int nb   = blockIdx.x / VG;
    const int vg   = blockIdx.x % VG;

    // ---- stage folded weights once per block (131072 B, L2-resident) ----
#if USE_ASYNC_LDS
    {
        for (int i = tid; i < 131072 / 16; i += 256) {
            gv4i* src = (gv4i*)((char*)gwqkv + i * 16);
            lv4i* dst = (lv4i*)(smem + SM_WQKV + i * 16);
            __builtin_amdgcn_global_load_async_to_lds_b128(src, dst, 0, 0);
        }
        __builtin_amdgcn_s_wait_asynccnt(0);
    }
#else
    {
        const v4u* s0 = (const v4u*)gwqkv;
        v4u* d0 = (v4u*)wqkv;
        for (int i = tid; i < 131072 / 16; i += 256) d0[i] = s0[i];
    }
#endif

    for (int it = 0; it < VG; it++) {
        const int vv = vg * VG + it;

        // ---- stage x slab: xs[t][c] = x[nb, c, t, vv] ----
        {
            const float* xb = x + (size_t)nb * CI * TT * VV + vv;
            for (int i = tid; i < CI * TT; i += 256) {
                int c = i >> 6, t = i & 63;
                xs[t * CI + c] = (_Float16)xb[(c * TT + t) * VV];
            }
        }
        __syncthreads();

        // ---- GEMM1: qkv[t][o] = xs (64x128) @ wqkv^T (128x384) ----
        for (int tile = wave; tile < 96; tile += 8) {
            int mt = tile / 24, nt = tile % 24;
            v8f acc = {};
#pragma unroll
            for (int kt = 0; kt < 4; kt++) {
                v16h a = frag_ld(xs   + (mt * 16 + lm) * CI + kt * 32 + lhi * 8);
                v16h b = frag_ld(wqkv + (nt * 16 + lm) * CI + kt * 32 + lhi * 8);
                acc = __builtin_amdgcn_wmma_f32_16x16x32_f16(false, a, false, b,
                                                             (short)0, acc, false, false);
            }
            int o = nt * 16 + lm;
            float bias = bqkv[o];
            // branchless routing: half-index into smem
            int qo = o & 127, h = qo >> 5, d = qo & 31;
            bool isv = (o >= 256);
            int base_qk = ((o < 128) ? (SM_Q / 2) : (SM_K / 2)) + (h * TT) * DD + d; // + t*DD
            int base_v  = (SM_VT / 2) + (h * DD + d) * TT;                           // + t
            union { v8f v; float f[8]; } u; u.v = acc;
#pragma unroll
            for (int r = 0; r < 8; r++) {
                int t = mt * 16 + lhi * 8 + r;
                int idx = isv ? (base_v + t) : (base_qk + t * DD);
                smh[idx] = (_Float16)(u.f[r] + bias);
            }
        }
        __syncthreads();

        // ---- cosine-norm q (fold logit scale and 1/sqrt(D)), norm k ----
        {
            int h = tid >> 6, t = tid & 63;
            float scale = expf(fminf(logit_scale[h], 4.6051702f));   // min(ls, log 100)
            union { v4u u[4]; _Float16 h16[32]; } rw;
            v4u* qp = (v4u*)(qbuf + (h * TT + t) * DD);
#pragma unroll
            for (int w = 0; w < 4; w++) rw.u[w] = qp[w];
            float ss = 0.f;
#pragma unroll
            for (int d = 0; d < DD; d++) { float q = (float)rw.h16[d]; ss += q * q; }
            float s = scale * 0.1767766953f / fmaxf(sqrtf(ss), 1e-12f); // * 1/sqrt(32)
#pragma unroll
            for (int d = 0; d < DD; d++) rw.h16[d] = (_Float16)((float)rw.h16[d] * s);
#pragma unroll
            for (int w = 0; w < 4; w++) qp[w] = rw.u[w];

            v4u* kp = (v4u*)(kbuf + (h * TT + t) * DD);
#pragma unroll
            for (int w = 0; w < 4; w++) rw.u[w] = kp[w];
            ss = 0.f;
#pragma unroll
            for (int d = 0; d < DD; d++) { float k = (float)rw.h16[d]; ss += k * k; }
            s = 1.f / fmaxf(sqrtf(ss), 1e-12f);
#pragma unroll
            for (int d = 0; d < DD; d++) rw.h16[d] = (_Float16)((float)rw.h16[d] * s);
#pragma unroll
            for (int w = 0; w < 4; w++) kp[w] = rw.u[w];
        }
        __syncthreads();

        // ---- GEMM2: scores[h] = q (64x32) @ k^T (32x64) ----
        for (int tile = wave; tile < 64; tile += 8) {
            int h = tile >> 4, mt = (tile >> 2) & 3, nt = tile & 3;
            v16h a = frag_ld(qbuf + (h * TT + mt * 16 + lm) * DD + lhi * 8);
            v16h b = frag_ld(kbuf + (h * TT + nt * 16 + lm) * DD + lhi * 8);
            v8f acc = {};
            acc = __builtin_amdgcn_wmma_f32_16x16x32_f16(false, a, false, b,
                                                         (short)0, acc, false, false);
            union { v8f v; float f[8]; } u; u.v = acc;
#pragma unroll
            for (int r = 0; r < 8; r++)
                sbuf[(h * TT + mt * 16 + lhi * 8 + r) * TT + nt * 16 + lm] = u.f[r];
        }
        __syncthreads();

        // ---- windowed softmax (|i-j|<=4) + rel bias; pack attn f16 in place ----
        {
            int h = tid >> 6, i = tid & 63;
            float* row = sbuf + (h * TT + i) * TT;
            _Float16* rowh = (_Float16*)row;          // 128-half row stride
            float e[9];
            float mx = -3.0e38f;
#pragma unroll
            for (int w = 0; w < 9; w++) {
                int j = i - 4 + w;
                bool ok = (j >= 0) && (j < TT);
                float sv = ok ? (row[j] + rel[h * 127 + (w + 59)]) : -3.0e38f;
                e[w] = sv;
                mx = fmaxf(mx, sv);
            }
            float sum = 0.f;
#pragma unroll
            for (int w = 0; w < 9; w++) { float t = expf(e[w] - mx); e[w] = t; sum += t; }
            float inv = 1.f / sum;
            // zero first 128B (attn region), then write the 9-tap window
            v4u z = {};
            v4u* rv = (v4u*)row;
#pragma unroll
            for (int w = 0; w < 8; w++) rv[w] = z;
#pragma unroll
            for (int w = 0; w < 9; w++) {
                int j = i - 4 + w;
                if (j >= 0 && j < TT) rowh[j] = (_Float16)(e[w] * inv);
            }
        }
        __syncthreads();

        // ---- GEMM3: out[h] = attn (64x64) @ val (64x32); val pre-transposed ----
        for (int tile = wave; tile < 32; tile += 8) {
            int h = tile >> 3, mt = (tile >> 1) & 3, nt = tile & 1;
            const _Float16* abase = (const _Float16*)(sbuf + h * TT * TT);
            v8f acc = {};
#pragma unroll
            for (int kt = 0; kt < 2; kt++) {
                v16h a = frag_ld(abase + (mt * 16 + lm) * 128 + kt * 32 + lhi * 8);
                v16h b = frag_ld(vbufT + (h * DD + nt * 16 + lm) * TT + kt * 32 + lhi * 8);
                acc = __builtin_amdgcn_wmma_f32_16x16x32_f16(false, a, false, b,
                                                             (short)0, acc, false, false);
            }
            union { v8f v; float f[8]; } u; u.v = acc;
#pragma unroll
            for (int r = 0; r < 8; r++)
                ao[(mt * 16 + lhi * 8 + r) * OUTC + h * DD + nt * 16 + lm] = (_Float16)u.f[r];
        }
        __syncthreads();

        // ---- GEMM4: proj + folded-BN bias + scaled residual -> global ----
        const size_t obase = (size_t)nb * OUTC * TT * VV + vv;
        for (int tile = wave; tile < 32; tile += 8) {
            int mt = tile >> 3, nt = tile & 7;
            v8f acc = {};
#pragma unroll
            for (int kt = 0; kt < 4; kt++) {
                v16h a = frag_ld(ao    + (mt * 16 + lm) * OUTC + kt * 32 + lhi * 8);
                v16h b = frag_ld(wproj + (nt * 16 + lm) * OUTC + kt * 32 + lhi * 8);
                acc = __builtin_amdgcn_wmma_f32_16x16x32_f16(false, a, false, b,
                                                             (short)0, acc, false, false);
            }
            int o2 = nt * 16 + lm;
            float bb = bproj[o2], rs = rscale[o2];
            union { v8f v; float f[8]; } u; u.v = acc;
#pragma unroll
            for (int r = 0; r < 8; r++) {
                int t = mt * 16 + lhi * 8 + r;
                size_t gi = obase + ((size_t)o2 * TT + t) * VV;
                out[gi] = u.f[r] + bb + rs * x[gi];   // residual in f32
            }
        }
        // next iteration only rewrites xs before its first barrier; safe.
    }
}

extern "C" void kernel_launch(void* const* d_in, const int* in_sizes, int n_in,
                              void* d_out, int out_size, void* d_ws, size_t ws_size,
                              hipStream_t stream) {
    const float* x         = (const float*)d_in[0];
    const float* bn_gamma  = (const float*)d_in[1];
    const float* bn_beta   = (const float*)d_in[2];
    const float* bn_mean   = (const float*)d_in[3];
    const float* bn_var    = (const float*)d_in[4];
    const float* qk_w      = (const float*)d_in[5];
    const float* qk_b      = (const float*)d_in[6];
    const float* v_w       = (const float*)d_in[7];
    const float* v_b       = (const float*)d_in[8];
    const float* logit_sc  = (const float*)d_in[9];
    const float* rel_bias  = (const float*)d_in[10];
    const float* proj_w    = (const float*)d_in[11];
    const float* proj_b    = (const float*)d_in[12];
    const float* bno_gamma = (const float*)d_in[13];
    const float* bno_beta  = (const float*)d_in[14];
    const float* bno_mean  = (const float*)d_in[15];
    const float* bno_var   = (const float*)d_in[16];

    char* ws = (char*)d_ws;
    _Float16* wqkv  = (_Float16*)(ws + WS_WQKV);
    _Float16* wproj = (_Float16*)(ws + WS_WPROJ);
    float* bqkv     = (float*)(ws + WS_BQKV);
    float* bproj    = (float*)(ws + WS_BPROJ);
    float* rscale   = (float*)(ws + WS_RSCALE);

    fold_kernel<<<2, 256, 0, stream>>>(bn_gamma, bn_beta, bn_mean, bn_var,
                                       qk_w, qk_b, v_w, v_b, proj_w, proj_b,
                                       bno_gamma, bno_beta, bno_mean, bno_var,
                                       wqkv, wproj, bqkv, bproj, rscale);

    attn_fused<<<NB * (VV / VG), 256, SM_TOTAL, stream>>>(
        x, logit_sc, rel_bias, wqkv, bqkv, bproj, rscale, (float*)d_out);
}
